// RecursiveMLP_44641890075293
// MI455X (gfx1250) — compile-verified, matrix operand order checked by reading, not run on hardware
//
#include <hip/hip_runtime.h>
#include <math.h>

// ---------- types ----------
typedef __bf16 bf16_t;
typedef bf16_t v16bf __attribute__((ext_vector_type(16)));
typedef float  v8f   __attribute__((ext_vector_type(8)));
typedef int    v2i   __attribute__((ext_vector_type(2)));

#define GAS __attribute__((address_space(1)))
#define LAS __attribute__((address_space(3)))

#define BATCH  8192
#define UNITS  512
#define INSZ   101
#define KPAD   128
#define HSTEPS 25
#define XCOLS  176

#if __has_builtin(__builtin_amdgcn_global_load_async_to_lds_b64) && \
    __has_builtin(__builtin_amdgcn_s_wait_asynccnt)
#define RM_ASYNC 1
#else
#define RM_ASYNC 0
#endif

__device__ __forceinline__ unsigned short f2bf(float f) {
    union { float f; unsigned int u; } a; a.f = f;
    unsigned int u = a.u;
    return (unsigned short)((u + 0x7FFFu + ((u >> 16) & 1u)) >> 16);
}
__device__ __forceinline__ float bf2f(unsigned short s) {
    union { unsigned int u; float f; } a; a.u = ((unsigned int)s) << 16; return a.f;
}
// branch-free tanh: copysign(1 - 2/(exp(2|x|)+1), x); exp->inf saturates to 1.
__device__ __forceinline__ float fast_tanh(float x) {
    float ax = fabsf(x);
    float e  = __expf(2.f * ax);
    float t  = 1.f - __fdividef(2.f, e + 1.f);
    return copysignf(t, x);
}
__device__ __forceinline__ void rm_wait_async() {
#if RM_ASYNC
    __builtin_amdgcn_s_wait_asynccnt(0);
#endif
}

union Frag { v16bf v; unsigned int u[8]; };

// ---------- weight packing into blocked B layout: Wp[((k>>5)*N + n)*32 + (k&31)] ----------
// forward weights: B[k][n] = W[k][n] (W is [Ksrc][N]), zero-pad k in [Ksrc, Kpad)
__global__ void rm_pack_blk(const float* __restrict__ W, unsigned short* __restrict__ out,
                            int Ksrc, int Kpad, int N) {
    int idx = blockIdx.x * 256 + threadIdx.x;
    if (idx >= Kpad * N) return;
    int k = idx / N, n = idx - k * N;
    float v = (k < Ksrc) ? W[(size_t)k * N + n] : 0.f;
    out[(((size_t)(k >> 5) * N + n) << 5) + (k & 31)] = f2bf(v);
}
// transposed weights: B[k][n] = W[n][k] (W is [Nsrc][Kc]), zero-pad n in [Nsrc, Nout)
__global__ void rm_pack_blkt(const float* __restrict__ W, unsigned short* __restrict__ out,
                             int Nsrc, int Kc, int Nout) {
    int idx = blockIdx.x * 256 + threadIdx.x;
    if (idx >= Kc * Nout) return;
    int k = idx / Nout, n = idx - k * Nout;
    float v = (n < Nsrc) ? W[(size_t)n * Kc + k] : 0.f;
    out[(((size_t)(k >> 5) * Nout + n) << 5) + (k & 31)] = f2bf(v);
}
__global__ void rm_pack_vec(const float* __restrict__ W, unsigned short* __restrict__ out, int n) {
    int idx = blockIdx.x * 256 + threadIdx.x;
    if (idx < n) out[idx] = f2bf(W[idx]);
}

// ---------- per-step data assembly: data[b][c] (bf16, padded to 128 cols) ----------
__global__ void rm_build_data(const float* __restrict__ x, const float* __restrict__ w,
                              unsigned short* __restrict__ out, int step) {
    int idx = blockIdx.x * 256 + threadIdx.x;       // BATCH*KPAD threads
    int b = idx >> 7;
    int c = idx & 127;
    int ne = HSTEPS - step;                          // 25 - i
    float v;
    if (c < ne)         v = x[(size_t)b * XCOLS + step + c];             // e[:, i:]
    else if (c < 25)    v = w[(size_t)b * HSTEPS + (c - ne)];            // w_cols[0:i]
    else if (c < 100)   v = x[(size_t)b * XCOLS + 26 + 3 * step + (c - 25)]; // v[:, i:i+25]
    else if (c == 100)  v = (float)step;
    else                v = 0.f;                                         // pad
    out[idx] = f2bf(v);
}

// ---------- WMMA GEMM: C[8192][N] = A[8192][K] @ B[K][N], B pre-blocked ----------
// MODE 0: Out_bf = tanh(acc + bias[n])
// MODE 1: Out_bf = acc * (1 - h[n]^2)          (tanh backward, Hin required)
// MODE 2: OutF[(row*25+step)*101 + n] = acc    (only n < 101)
template <int MODE>
__global__ __launch_bounds__(256) void rm_gemm_bf16(
    const unsigned short* __restrict__ A,
    const unsigned short* __restrict__ Wp,      // blocked: [(K/32)][N][32]
    const float* __restrict__ bias,
    const unsigned short* __restrict__ Hin,
    unsigned short* __restrict__ OutBf,
    float* __restrict__ OutF,
    int K, int N, int step)
{
    // double-buffered tiles; rows padded to 36 shorts (72B) for 8B-aligned b64 LDS ops
    __shared__ unsigned short sA[2][128][36];   // [m][k]
    __shared__ unsigned short sB[2][128][36];   // [n][k]

    const int tid  = threadIdx.x;
    const int wave = tid >> 5;
    const int lane = tid & 31;
    const int wr   = wave & 3;     // 4 waves along M (32 rows each)
    const int wc   = wave >> 2;    // 2 waves along N (64 cols each)
    const int mBase = blockIdx.y * 128;
    const int nBase = blockIdx.x * 128;
    const int half = lane >> 4;
    const int ml   = lane & 15;

    // staging geometry: 128 rows x 32 shorts per tile, 8B chunks, 256 thr x 4 chunks
    const int r0 = tid >> 3;             // base row 0..31 (+ c*32)
    const int c4 = (tid & 7) << 2;       // chunk offset in shorts

    auto stage = [&](int k0, int b) {
#if RM_ASYNC
        #pragma unroll
        for (int c = 0; c < 4; c++) {
            int row = r0 + c * 32;
            const unsigned short* gp = A + (size_t)(mBase + row) * K + k0 + c4;
            __builtin_amdgcn_global_load_async_to_lds_b64(
                (GAS v2i*)gp, (LAS v2i*)&sA[b][row][c4], 0, 0);
        }
        #pragma unroll
        for (int c = 0; c < 4; c++) {
            int row = r0 + c * 32;
            const unsigned short* gp =
                Wp + (((size_t)(k0 >> 5) * N + nBase + row) << 5) + c4;
            __builtin_amdgcn_global_load_async_to_lds_b64(
                (GAS v2i*)gp, (LAS v2i*)&sB[b][row][c4], 0, 0);
        }
#else
        // batched: issue ALL global loads first, then LDS stores (waits stagger)
        uint2 ra[4], rb[4];
        #pragma unroll
        for (int c = 0; c < 4; c++) {
            int row = r0 + c * 32;
            ra[c] = *(const uint2*)(A + (size_t)(mBase + row) * K + k0 + c4);
        }
        #pragma unroll
        for (int c = 0; c < 4; c++) {
            int row = r0 + c * 32;
            rb[c] = *(const uint2*)(Wp + (((size_t)(k0 >> 5) * N + nBase + row) << 5) + c4);
        }
        #pragma unroll
        for (int c = 0; c < 4; c++) *(uint2*)&sA[b][r0 + c * 32][c4] = ra[c];
        #pragma unroll
        for (int c = 0; c < 4; c++) *(uint2*)&sB[b][r0 + c * 32][c4] = rb[c];
#endif
    };

    v8f acc[2][4] = {};

    stage(0, 0);
    rm_wait_async();
    __syncthreads();

    const int nk = K >> 5;
    for (int ki = 0; ki < nk; ki++) {
        const int b = ki & 1;
        if (ki + 1 < nk) stage((ki + 1) << 5, b ^ 1);   // prefetch next tile

        // fragments per ISA bf16 A/B layouts (K pairs contiguous)
        Frag af[2];
        #pragma unroll
        for (int mt = 0; mt < 2; mt++) {
            int row = wr * 32 + mt * 16 + ml;
            #pragma unroll
            for (int j = 0; j < 8; j++) {
                int e  = 2 * j;
                int kk = (e < 8 ? e : e + 8) + 8 * half;
                af[mt].u[j] = *(const unsigned int*)&sA[b][row][kk];
            }
        }
        Frag bfr[4];
        #pragma unroll
        for (int nt = 0; nt < 4; nt++) {
            int coln = wc * 64 + nt * 16 + ml;
            #pragma unroll
            for (int j = 0; j < 8; j++) {
                int e  = 2 * j;
                int kk = (e < 8 ? e : e + 8) + 8 * half;
                bfr[nt].u[j] = *(const unsigned int*)&sB[b][coln][kk];
            }
        }
        #pragma unroll
        for (int mt = 0; mt < 2; mt++)
            #pragma unroll
            for (int nt = 0; nt < 4; nt++)
                acc[mt][nt] = __builtin_amdgcn_wmma_f32_16x16x32_bf16(
                    false, af[mt].v, false, bfr[nt].v,
                    (short)0, acc[mt][nt], false, false);

        rm_wait_async();     // staged tile b^1 fully in LDS
        __syncthreads();
    }

    // --- fused epilogue (straight-line per MODE; per-column values hoisted) ---
    #pragma unroll
    for (int mt = 0; mt < 2; mt++) {
        const int rowB = mBase + wr * 32 + mt * 16 + 8 * half;
        #pragma unroll
        for (int nt = 0; nt < 4; nt++) {
            const int coln = nBase + wc * 64 + nt * 16 + ml;
            if (MODE == 0) {
                const float bb = bias[coln];
                #pragma unroll
                for (int r = 0; r < 8; r++) {
                    int row = rowB + r;
                    OutBf[(size_t)row * N + coln] = f2bf(fast_tanh(acc[mt][nt][r] + bb));
                }
            } else if (MODE == 1) {
                const unsigned short* hp = Hin + coln;
                #pragma unroll
                for (int r = 0; r < 8; r++) {
                    int row = rowB + r;
                    float h = bf2f(hp[(size_t)row * N]);
                    OutBf[(size_t)row * N + coln] = f2bf(acc[mt][nt][r] * (1.f - h * h));
                }
            } else {
                if (coln < INSZ) {
                    float* gp = OutF + (size_t)step * INSZ + coln;
                    #pragma unroll
                    for (int r = 0; r < 8; r++) {
                        int row = rowB + r;
                        gp[(size_t)row * HSTEPS * INSZ] = acc[mt][nt][r];
                    }
                }
            }
        }
    }
}

// ---------- head GEMV: w[b,step] = dot(h3[b], W4) + b4 (wave-per-row) ----------
__global__ __launch_bounds__(256) void rm_head_gemv(
    const unsigned short* __restrict__ h3, const unsigned short* __restrict__ W4b,
    const float* __restrict__ b4, float* __restrict__ wOut, int step)
{
    int wave = threadIdx.x >> 5, lane = threadIdx.x & 31;
    int row = blockIdx.x * 8 + wave;
    const unsigned short* hp = h3 + (size_t)row * UNITS;
    float acc = 0.f;
    #pragma unroll
    for (int j = 0; j < 16; j++) {
        int idx = j * 32 + lane;
        acc += bf2f(hp[idx]) * bf2f(W4b[idx]);
    }
    #pragma unroll
    for (int off = 16; off > 0; off >>= 1) acc += __shfl_xor(acc, off, 32);
    if (lane == 0) wOut[(size_t)row * HSTEPS + step] = acc + b4[0];
}

// ---------- dz3 = W4[n] * (1 - h3^2) ----------
__global__ void rm_dz_head(const float* __restrict__ W4, const unsigned short* __restrict__ h3,
                           unsigned short* __restrict__ dz) {
    int idx = blockIdx.x * 256 + threadIdx.x;    // BATCH*UNITS
    int n = idx & (UNITS - 1);
    float h = bf2f(h3[idx]);
    dz[idx] = f2bf(W4[n] * (1.f - h * h));
}

// ---------- host ----------
extern "C" void kernel_launch(void* const* d_in, const int* in_sizes, int n_in,
                              void* d_out, int out_size, void* d_ws, size_t ws_size,
                              hipStream_t stream) {
    (void)in_sizes; (void)n_in; (void)out_size; (void)ws_size;
    const float* x  = (const float*)d_in[0];
    const float* W0 = (const float*)d_in[1];  const float* b0 = (const float*)d_in[2];
    const float* W1 = (const float*)d_in[3];  const float* b1 = (const float*)d_in[4];
    const float* W2 = (const float*)d_in[5];  const float* b2 = (const float*)d_in[6];
    const float* W3 = (const float*)d_in[7];  const float* b3 = (const float*)d_in[8];
    const float* W4 = (const float*)d_in[9];  const float* b4 = (const float*)d_in[10];

    float* wOut = (float*)d_out;                                   // (B,25)
    float* gOut = (float*)d_out + (size_t)BATCH * HSTEPS;          // (B,25,101)

    // workspace carve-out (256B aligned)
    char* ws = (char*)d_ws;
    size_t off = 0;
    auto carve = [&](size_t bytes) -> char* {
        char* p = ws + off;
        off = (off + bytes + 255) & ~(size_t)255;
        return p;
    };
    unsigned short* W0b = (unsigned short*)carve((size_t)KPAD * UNITS * 2);
    unsigned short* W1b = (unsigned short*)carve((size_t)UNITS * UNITS * 2);
    unsigned short* W2b = (unsigned short*)carve((size_t)UNITS * UNITS * 2);
    unsigned short* W3b = (unsigned short*)carve((size_t)UNITS * UNITS * 2);
    unsigned short* W4b = (unsigned short*)carve((size_t)UNITS * 2);
    unsigned short* W1t = (unsigned short*)carve((size_t)UNITS * UNITS * 2);
    unsigned short* W2t = (unsigned short*)carve((size_t)UNITS * UNITS * 2);
    unsigned short* W3t = (unsigned short*)carve((size_t)UNITS * UNITS * 2);
    unsigned short* W0t = (unsigned short*)carve((size_t)UNITS * KPAD * 2);
    unsigned short* dat = (unsigned short*)carve((size_t)BATCH * KPAD * 2);
    unsigned short* h0  = (unsigned short*)carve((size_t)BATCH * UNITS * 2);
    unsigned short* h1  = (unsigned short*)carve((size_t)BATCH * UNITS * 2);
    unsigned short* h2  = (unsigned short*)carve((size_t)BATCH * UNITS * 2);
    unsigned short* h3  = (unsigned short*)carve((size_t)BATCH * UNITS * 2);
    unsigned short* dzA = (unsigned short*)carve((size_t)BATCH * UNITS * 2);
    unsigned short* dzB = (unsigned short*)carve((size_t)BATCH * UNITS * 2);

    // pack weights to blocked bf16 (+ transposed copies for backward)
    {
        int n;
        n = KPAD * UNITS;  rm_pack_blk <<<(n + 255) / 256, 256, 0, stream>>>(W0, W0b, INSZ, KPAD, UNITS);
        n = UNITS * UNITS; rm_pack_blk <<<(n + 255) / 256, 256, 0, stream>>>(W1, W1b, UNITS, UNITS, UNITS);
        n = UNITS * UNITS; rm_pack_blk <<<(n + 255) / 256, 256, 0, stream>>>(W2, W2b, UNITS, UNITS, UNITS);
        n = UNITS * UNITS; rm_pack_blk <<<(n + 255) / 256, 256, 0, stream>>>(W3, W3b, UNITS, UNITS, UNITS);
        n = UNITS;         rm_pack_vec <<<(n + 255) / 256, 256, 0, stream>>>(W4, W4b, UNITS);
        n = UNITS * UNITS; rm_pack_blkt<<<(n + 255) / 256, 256, 0, stream>>>(W1, W1t, UNITS, UNITS, UNITS);
        n = UNITS * UNITS; rm_pack_blkt<<<(n + 255) / 256, 256, 0, stream>>>(W2, W2t, UNITS, UNITS, UNITS);
        n = UNITS * UNITS; rm_pack_blkt<<<(n + 255) / 256, 256, 0, stream>>>(W3, W3t, UNITS, UNITS, UNITS);
        n = UNITS * KPAD;  rm_pack_blkt<<<(n + 255) / 256, 256, 0, stream>>>(W0, W0t, INSZ, UNITS, KPAD);
    }

    const dim3 blk(256);
    const dim3 gFull(UNITS / 128, BATCH / 128);   // (4, 64)
    const dim3 gPad (KPAD  / 128, BATCH / 128);   // (1, 64)
    const int nBuild = (BATCH * KPAD) / 256;
    const int nElem  = (BATCH * UNITS) / 256;

    for (int i = 0; i < HSTEPS; i++) {
        // data assembly (reads w columns produced by earlier steps)
        rm_build_data<<<nBuild, blk, 0, stream>>>(x, wOut, dat, i);
        // forward
        rm_gemm_bf16<0><<<gFull, blk, 0, stream>>>(dat, W0b, b0, nullptr, h0, nullptr, KPAD,  UNITS, i);
        rm_gemm_bf16<0><<<gFull, blk, 0, stream>>>(h0,  W1b, b1, nullptr, h1, nullptr, UNITS, UNITS, i);
        rm_gemm_bf16<0><<<gFull, blk, 0, stream>>>(h1,  W2b, b2, nullptr, h2, nullptr, UNITS, UNITS, i);
        rm_gemm_bf16<0><<<gFull, blk, 0, stream>>>(h2,  W3b, b3, nullptr, h3, nullptr, UNITS, UNITS, i);
        rm_head_gemv<<<BATCH / 8, blk, 0, stream>>>(h3, W4b, b4, wOut, i);
        // backward (VJP w.r.t. data)
        rm_dz_head<<<nElem, blk, 0, stream>>>(W4, h3, dzA);
        rm_gemm_bf16<1><<<gFull, blk, 0, stream>>>(dzA, W3t, nullptr, h2, dzB, nullptr, UNITS, UNITS, i);
        rm_gemm_bf16<1><<<gFull, blk, 0, stream>>>(dzB, W2t, nullptr, h1, dzA, nullptr, UNITS, UNITS, i);
        rm_gemm_bf16<1><<<gFull, blk, 0, stream>>>(dzA, W1t, nullptr, h0, dzB, nullptr, UNITS, UNITS, i);
        rm_gemm_bf16<2><<<gPad,  blk, 0, stream>>>(dzB, W0t, nullptr, nullptr, nullptr, gOut, UNITS, KPAD, i);
    }
}
